// BitLinear_60713657696634
// MI455X (gfx1250) — 8-bit matrix operand order repaired and checked by reading, compiled, not run on hardware
//
#include <hip/hip_runtime.h>
#include <stdint.h>

#define BL_EPS 1e-5f
#define K_DIM 4096
#define N_DIM 4096
#define M_DIM 8192
#define W_ELEMS (4096.0f * 4096.0f)

typedef __attribute__((ext_vector_type(8))) int v8i;

__global__ __launch_bounds__(256)
void bl_abs_sum(const float* __restrict__ w, float* __restrict__ sum_out, int n4) {
    const float4* w4 = reinterpret_cast<const float4*>(w);
    float acc = 0.0f;
    for (int i = blockIdx.x * blockDim.x + threadIdx.x; i < n4;
         i += gridDim.x * blockDim.x) {
        float4 v = w4[i];
        acc += fabsf(v.x) + fabsf(v.y) + fabsf(v.z) + fabsf(v.w);
    }
    __shared__ float sm[256];
    sm[threadIdx.x] = acc;
    __syncthreads();
    for (int s = 128; s > 0; s >>= 1) {
        if ((int)threadIdx.x < s) sm[threadIdx.x] += sm[threadIdx.x + s];
        __syncthreads();
    }
    if (threadIdx.x == 0) atomicAdd(sum_out, sm[0]);
}

__global__ __launch_bounds__(256)
void bl_wquant(const float* __restrict__ w, const float* __restrict__ sum_in,
               int8_t* __restrict__ wq, int n4) {
    const float delta = (*sum_in) * (1.0f / W_ELEMS) + BL_EPS;
    const float inv = 1.0f / delta;
    const float4* w4 = reinterpret_cast<const float4*>(w);
    char4* q4 = reinterpret_cast<char4*>(wq);
    for (int i = blockIdx.x * blockDim.x + threadIdx.x; i < n4;
         i += gridDim.x * blockDim.x) {
        float4 v = w4[i];
        char4 o;
        o.x = (signed char)fmaxf(-1.0f, fminf(1.0f, rintf(v.x * inv)));
        o.y = (signed char)fmaxf(-1.0f, fminf(1.0f, rintf(v.y * inv)));
        o.z = (signed char)fmaxf(-1.0f, fminf(1.0f, rintf(v.z * inv)));
        o.w = (signed char)fmaxf(-1.0f, fminf(1.0f, rintf(v.w * inv)));
        q4[i] = o;
    }
}

__global__ __launch_bounds__(256)
void bl_xquant(const float* __restrict__ x, int8_t* __restrict__ xq) {
    const int row = blockIdx.x;
    const float4* xr = reinterpret_cast<const float4*>(x + (size_t)row * K_DIM);
    char4* qr = reinterpret_cast<char4*>(xq + (size_t)row * K_DIM);

    float4 v[4];
    float m = 0.0f;
#pragma unroll
    for (int i = 0; i < 4; ++i) {
        v[i] = xr[threadIdx.x + 256 * i];
        m = fmaxf(m, fabsf(v[i].x));
        m = fmaxf(m, fabsf(v[i].y));
        m = fmaxf(m, fabsf(v[i].z));
        m = fmaxf(m, fabsf(v[i].w));
    }
    __shared__ float sm[256];
    sm[threadIdx.x] = m;
    __syncthreads();
    for (int s = 128; s > 0; s >>= 1) {
        if ((int)threadIdx.x < s)
            sm[threadIdx.x] = fmaxf(sm[threadIdx.x], sm[threadIdx.x + s]);
        __syncthreads();
    }
    const float gamma = sm[0] + BL_EPS;
    const float sc = 127.0f / gamma;
#pragma unroll
    for (int i = 0; i < 4; ++i) {
        char4 o;
        o.x = (signed char)fmaxf(-128.0f, fminf(127.0f, rintf(v[i].x * sc)));
        o.y = (signed char)fmaxf(-128.0f, fminf(127.0f, rintf(v[i].y * sc)));
        o.z = (signed char)fmaxf(-128.0f, fminf(127.0f, rintf(v[i].z * sc)));
        o.w = (signed char)fmaxf(-128.0f, fminf(127.0f, rintf(v[i].w * sc)));
        qr[threadIdx.x + 256 * i] = o;
    }
}

__global__ __launch_bounds__(256)
void bl_gemm_iu8(const int8_t* __restrict__ aq, const int8_t* __restrict__ wq,
                 const float* __restrict__ sum_in, float* __restrict__ out) {
    const int lane = threadIdx.x & 31;
    const int wave = threadIdx.x >> 5;
    const int waveM = wave & 3;
    const int waveN = wave >> 2;
    const int mBase = blockIdx.y * 128 + waveM * 32;
    const int nBase = blockIdx.x * 128 + waveN * 64;
    const int l16 = lane & 15;
    const int hi = lane >> 4;

    v8i acc[2][4] = {};

    const int8_t* aPtr0 = aq + (size_t)(mBase + l16) * K_DIM + hi * 16;
    const int8_t* aPtr1 = aPtr0 + (size_t)16 * K_DIM;
    const int8_t* bPtr[4];
#pragma unroll
    for (int nt = 0; nt < 4; ++nt)
        bPtr[nt] = wq + (size_t)(nBase + nt * 16 + l16) * K_DIM + hi * 16;

    for (int k0 = 0; k0 < K_DIM; k0 += 64) {
        v8i a[2], b[4];
#pragma unroll
        for (int mt = 0; mt < 2; ++mt) {
            const int8_t* p = (mt == 0 ? aPtr0 : aPtr1) + k0;
            int2 c0 = *reinterpret_cast<const int2*>(p);
            int2 c1 = *reinterpret_cast<const int2*>(p + 8);
            int2 c2 = *reinterpret_cast<const int2*>(p + 32);
            int2 c3 = *reinterpret_cast<const int2*>(p + 40);
            a[mt][0] = c0.x; a[mt][1] = c0.y;
            a[mt][2] = c1.x; a[mt][3] = c1.y;
            a[mt][4] = c2.x; a[mt][5] = c2.y;
            a[mt][6] = c3.x; a[mt][7] = c3.y;
        }
#pragma unroll
        for (int nt = 0; nt < 4; ++nt) {
            const int8_t* p = bPtr[nt] + k0;
            int4 c0 = *reinterpret_cast<const int4*>(p);
            int4 c1 = *reinterpret_cast<const int4*>(p + 32);
            b[nt][0] = c0.x; b[nt][1] = c0.y; b[nt][2] = c0.z; b[nt][3] = c0.w;
            b[nt][4] = c1.x; b[nt][5] = c1.y; b[nt][6] = c1.z; b[nt][7] = c1.w;
        }
        if (k0 + 64 < K_DIM) {
            __builtin_prefetch(aPtr0 + k0 + 64, 0, 3);
            __builtin_prefetch(aPtr1 + k0 + 64, 0, 3);
            __builtin_prefetch(bPtr[0] + k0 + 64, 0, 3);
            __builtin_prefetch(bPtr[2] + k0 + 64, 0, 3);
        }
#pragma unroll
        for (int mt = 0; mt < 2; ++mt)
#pragma unroll
            for (int nt = 0; nt < 4; ++nt)
                acc[mt][nt] = __builtin_amdgcn_wmma_i32_16x16x64_iu8(
                     true, a[mt],  true, b[nt],
                    acc[mt][nt],  false,  false);
    }

    const float scale =
        ((*sum_in) * (1.0f / W_ELEMS) + BL_EPS) * (1.0f / 127.0f);
#pragma unroll
    for (int mt = 0; mt < 2; ++mt)
#pragma unroll
        for (int nt = 0; nt < 4; ++nt)
#pragma unroll
            for (int j = 0; j < 8; ++j) {
                const int m = mBase + mt * 16 + j + hi * 8;
                const int n = nBase + nt * 16 + l16;
                out[(size_t)m * N_DIM + n] = scale * (float)acc[mt][nt][j];
            }
}

extern "C" void kernel_launch(void* const* d_in, const int* in_sizes, int n_in,
                              void* d_out, int out_size, void* d_ws, size_t ws_size,
                              hipStream_t stream) {
    const float* x = (const float*)d_in[0];
    const float* w = (const float*)d_in[1];
    float* out = (float*)d_out;

    uint8_t* ws = (uint8_t*)d_ws;
    float* d_sum = (float*)ws;
    int8_t* d_wq = (int8_t*)(ws + 64);
    int8_t* d_xq = (int8_t*)(ws + 64 + (size_t)N_DIM * K_DIM);

    hipMemsetAsync(d_sum, 0, sizeof(float), stream);

    const int n4w = (N_DIM * K_DIM) / 4;
    bl_abs_sum<<<1024, 256, 0, stream>>>(w, d_sum, n4w);
    bl_wquant<<<1024, 256, 0, stream>>>(w, d_sum, d_wq, n4w);
    bl_xquant<<<M_DIM, 256, 0, stream>>>(x, d_xq);

    dim3 grid(N_DIM / 128, M_DIM / 128);
    bl_gemm_iu8<<<grid, 256, 0, stream>>>(d_xq, d_wq, d_sum, out);
}
